// GNNLayer_33492154974250
// MI455X (gfx1250) — matrix operand order repaired and checked by reading, not compiled, not run on hardware
//
#include <hip/hip_runtime.h>

#define N_NODES 50000
#define N_EDGES 800000
#define D 128
#define NEG_SLOPE 0.01f
#define BLOCK 256
#define WAVES 8
#define MTILE (WAVES * 16)   // 128 node rows per block

typedef __attribute__((ext_vector_type(16))) __bf16 v16bf;
typedef __attribute__((ext_vector_type(8)))  __bf16 v8bf;
typedef __attribute__((ext_vector_type(4)))  __bf16 v4bf;
typedef __attribute__((ext_vector_type(8)))  float  v8f;

// ----------------------------------------------------- zero agg (bf16 buffer)
__global__ void gnn_zero_kernel(unsigned int* __restrict__ p, int nDwords) {
    int i = blockIdx.x * blockDim.x + threadIdx.x;
    if (i < nDwords) p[i] = 0u;
}

// ----------------------------------------------------- x: f32 -> bf16 once
__global__ void gnn_cvtx_kernel(const float* __restrict__ x,
                                __bf16* __restrict__ xb, int n4) {
    int i = blockIdx.x * blockDim.x + threadIdx.x;
    if (i >= n4) return;
    float4 f = reinterpret_cast<const float4*>(x)[i];
    v4bf o;
    o[0] = (__bf16)f.x; o[1] = (__bf16)f.y; o[2] = (__bf16)f.z; o[3] = (__bf16)f.w;
    *reinterpret_cast<v4bf*>(xb + (long long)i * 4) = o;
}

// ------------------------------------------------- edge gather + scatter-add
// One wave per edge: 32 lanes x 4 bf16 dims = full 128-dim row (256 B, L2-hit).
// GLOBAL_ATOMIC_PK_ADD_BF16: 2 values per atomic -> 51.2M atomics instead of
// 102.4M f32 atomics (phase is L2-atomic-throughput-bound).
__global__ void gnn_scatter_kernel(const __bf16* __restrict__ xb,
                                   const int* __restrict__ src,
                                   const int* __restrict__ dst,
                                   __bf16* __restrict__ agg) {
    long long t = (long long)blockIdx.x * blockDim.x + threadIdx.x;
    int e = (int)(t >> 5);
    if (e >= N_EDGES) return;
    int c = (int)(t & 31) << 2;                       // 4 dims per lane
    int s = src[e];
    int d = dst[e];
    uint2 v = *reinterpret_cast<const uint2*>(xb + (long long)s * D + c);
    unsigned long long addr =
        (unsigned long long)(agg + (long long)d * D + c);
    asm volatile("global_atomic_pk_add_bf16 %0, %1, off"
                 :: "v"(addr), "v"(v.x) : "memory");
    asm volatile("global_atomic_pk_add_bf16 %0, %1, off offset:4"
                 :: "v"(addr), "v"(v.y) : "memory");
}

// ---------------------------------------------------------- fragment loaders
// A-matrix 16x32 bf16: lane<16 -> row=lane,  K = kt*32 + 0..15
//                      lane>=16 -> row=lane-16, K = kt*32 + 16..31
__device__ __forceinline__ v16bf load_a_bf16(const __bf16* __restrict__ row,
                                             int kt, int lane) {
    int kbase = kt * 32 + ((lane & 16) ? 16 : 0);
    v8bf lo = *(const v8bf*)(row + kbase);
    v8bf hi = *(const v8bf*)(row + kbase + 8);
    return __builtin_shufflevector(lo, hi, 0, 1, 2, 3, 4, 5, 6, 7,
                                           8, 9, 10, 11, 12, 13, 14, 15);
}

// B-matrix 32x16 bf16: lane<16 -> col n=lane,  K = kt*32 + 0..15
//                      lane>=16 -> col n=lane-16, K = kt*32 + 16..31
// B[k][n] == W[n][k] (W row-major [N][K]) -> 16 consecutive K per lane.
__device__ __forceinline__ v16bf load_b_lds(const __bf16* w, int ncol, int kt, int lane) {
    int koff = kt * 32 + ((lane & 16) ? 16 : 0);
    return *(const v16bf*)(w + ncol * D + koff);
}

// --------------------------------------------- fused GEMM1 + LeakyReLU + GEMM2
__global__ __launch_bounds__(BLOCK)
void gnn_gemm_kernel(const __bf16* __restrict__ xb,
                     const __bf16* __restrict__ agg,
                     const float* __restrict__ W_rel,
                     const float* __restrict__ b_rel,
                     const float* __restrict__ W_root,
                     const float* __restrict__ W_lin,
                     const float* __restrict__ b_lin,
                     float* __restrict__ out) {
    extern __shared__ char smem[];
    __bf16* w_rel  = (__bf16*)smem;          // 128x128 bf16 = 32 KB
    __bf16* w_root = w_rel + D * D;          // 32 KB
    __bf16* w_lin  = w_root + D * D;         // 32 KB
    __bf16* h_lds  = w_lin + D * D;          // 8 waves * 16 * 128 bf16 = 32 KB

    // Stage weights to LDS as bf16 (block-cooperative).
    for (int i = threadIdx.x; i < D * D; i += BLOCK) {
        w_rel[i]  = (__bf16)W_rel[i];
        w_root[i] = (__bf16)W_root[i];
        w_lin[i]  = (__bf16)W_lin[i];
    }
    __syncthreads();

    const int wave = threadIdx.x >> 5;
    const int lane = threadIdx.x & 31;
    const int mbase = blockIdx.x * MTILE + wave * 16;

    // Per-lane A row (clamped; WMMA requires EXEC all-ones, mask only at store).
    int arow = mbase + (lane & 15);
    if (arow >= N_NODES) arow = N_NODES - 1;
    const __bf16* agg_row = agg + (long long)arow * D;
    const __bf16* x_row   = xb  + (long long)arow * D;

    const int nlane = lane & 15;
    const int mofs  = (lane & 16) ? 8 : 0;

    // ---- GEMM1: h = agg @ W_rel^T + x @ W_root^T + b_rel; LeakyReLU -> LDS
    #pragma unroll
    for (int nt = 0; nt < 8; ++nt) {
        v8f c = {};
        #pragma unroll
        for (int kt = 0; kt < 4; ++kt) {
            v16bf a0 = load_a_bf16(agg_row, kt, lane);
            v16bf b0 = load_b_lds(w_rel, nt * 16 + nlane, kt, lane);
            c = __builtin_amdgcn_wmma_f32_16x16x32_bf16(false, a0, false, b0,
                                                        (short)0, c, false, false);
            v16bf a1 = load_a_bf16(x_row, kt, lane);
            v16bf b1 = load_b_lds(w_root, nt * 16 + nlane, kt, lane);
            c = __builtin_amdgcn_wmma_f32_16x16x32_bf16(false, a1, false, b1,
                                                        (short)0, c, false, false);
        }
        const int n = nt * 16 + nlane;
        const float bias = b_rel[n];
        // C layout: VGPR r -> M = r + mofs, N = nlane (ISA 7.12.2)
        #pragma unroll
        for (int r = 0; r < 8; ++r) {
            float hv = c[r] + bias;
            hv = (hv >= 0.0f) ? hv : NEG_SLOPE * hv;
            h_lds[(wave * 16 + mofs + r) * D + n] = (__bf16)hv;
        }
    }
    __syncthreads();

    // ---- GEMM2: out = h @ W_lin^T + b_lin
    const __bf16* hbase = h_lds + wave * 16 * D;
    #pragma unroll
    for (int nt = 0; nt < 8; ++nt) {
        v8f c = {};
        #pragma unroll
        for (int kt = 0; kt < 4; ++kt) {
            v16bf a = load_a_bf16(hbase + nlane * D, kt, lane);
            v16bf b = load_b_lds(w_lin, nt * 16 + nlane, kt, lane);
            c = __builtin_amdgcn_wmma_f32_16x16x32_bf16(false, a, false, b,
                                                        (short)0, c, false, false);
        }
        const int n = nt * 16 + nlane;
        const float bias = b_lin[n];
        #pragma unroll
        for (int r = 0; r < 8; ++r) {
            const int row = mbase + mofs + r;
            if (row < N_NODES)
                out[(long long)row * D + n] = c[r] + bias;
        }
    }
}

// ------------------------------------------------------------------ launcher
extern "C" void kernel_launch(void* const* d_in, const int* in_sizes, int n_in,
                              void* d_out, int out_size, void* d_ws, size_t ws_size,
                              hipStream_t stream) {
    const float* x      = (const float*)d_in[0];
    const int*   ei     = (const int*)d_in[1];   // [2, E]: src row then dst row
    const float* W_rel  = (const float*)d_in[2];
    const float* b_rel  = (const float*)d_in[3];
    const float* W_root = (const float*)d_in[4];
    const float* W_lin  = (const float*)d_in[5];
    const float* W_lin_b= (const float*)d_in[6];
    float* out = (float*)d_out;

    // Workspace: [agg bf16: N*D*2 B = 12.8 MB][x bf16: N*D*2 B = 12.8 MB]
    __bf16* agg = (__bf16*)d_ws;
    __bf16* xb  = (__bf16*)((char*)d_ws + (size_t)N_NODES * D * 2);

    const int nDw = N_NODES * D / 2;              // agg dwords to zero
    gnn_zero_kernel<<<(nDw + BLOCK - 1) / BLOCK, BLOCK, 0, stream>>>(
        (unsigned int*)agg, nDw);

    const int n4 = N_NODES * D / 4;               // float4 chunks of x
    gnn_cvtx_kernel<<<(n4 + BLOCK - 1) / BLOCK, BLOCK, 0, stream>>>(x, xb, n4);

    const long long tScatter = (long long)N_EDGES * 32;
    gnn_scatter_kernel<<<(unsigned)((tScatter + BLOCK - 1) / BLOCK), BLOCK, 0, stream>>>(
        xb, ei, ei + N_EDGES, agg);

    const int nBlocks = (N_NODES + MTILE - 1) / MTILE;
    const size_t shmem = (size_t)(3 * D * D + WAVES * 16 * D) * 2;  // 128 KB LDS
    gnn_gemm_kernel<<<nBlocks, BLOCK, shmem, stream>>>(
        xb, agg, W_rel, b_rel, W_root, W_lin, W_lin_b, out);
}
